// Model_50508815401322
// MI455X (gfx1250) — matrix operand order repaired and checked by reading, not compiled, not run on hardware
//
#include <hip/hip_runtime.h>

// ---------------------------------------------------------------------------
// Types for CDNA5 WMMA
// ---------------------------------------------------------------------------
typedef __bf16 v16bf __attribute__((ext_vector_type(16)));
typedef __bf16 v8bf  __attribute__((ext_vector_type(8)));
typedef __bf16 v2bf  __attribute__((ext_vector_type(2)));
typedef float  v8f   __attribute__((ext_vector_type(8)));

// native RNE f32->bf16 pair convert + single 32-bit LDS store
static __device__ __forceinline__ void st_bf2(__bf16* dst, float a, float b) {
    v2bf p;
    p[0] = (__bf16)a;
    p[1] = (__bf16)b;
    *(v2bf*)dst = p;
}

static __device__ __forceinline__ v16bf frag16(const __bf16* p0, const __bf16* p1) {
    v8bf lo = *(const v8bf*)p0;
    v8bf hi = *(const v8bf*)p1;
    v16bf r;
#pragma unroll
    for (int i = 0; i < 8; ++i) { r[i] = lo[i]; r[i + 8] = hi[i]; }
    return r;
}

// ---- CDNA5 async global->LDS copy (ASYNCcnt-tracked, bypasses VGPRs) ------
static __device__ __forceinline__ void async_b128(void* lds, const void* g) {
    unsigned l = (unsigned)(size_t)lds;                 // low 32 bits = LDS offset
    unsigned long long a = (unsigned long long)(size_t)g;
    asm volatile("global_load_async_to_lds_b128 %0, %1, off" :: "v"(l), "v"(a) : "memory");
}
static __device__ __forceinline__ void async_wait0() {
    asm volatile("s_wait_asynccnt 0x0" ::: "memory");
}

// ---------------------------------------------------------------------------
// Model constants
// ---------------------------------------------------------------------------
#define Dm   768
#define Hh   12
#define DH   64
#define FFm  3072
#define Ss   4096
#define Tt   128
#define Ww   256
#define Vv   50265
#define NC   16          // S / W
#define PADID 1

// ---------------------------------------------------------------------------
// GEMM:  C[M,N] = A[M,K] @ W + bias   (A fp32 row-major; W fp32;
//        BT=0: W is [K,N] row-major; BT=1: W is [N,K] row-major)
//        GELU=1: exact-erf gelu epilogue. FULL=1: M%128==0 && N%128==0.
// Block tile 128x128x32, 256 threads = 8 waves (2 M-waves x 4 N-waves),
// wave tile 64x32 = 8 v_wmma_f32_16x16x32_bf16 per K-step.
// Software-pipelined: next tile's global loads issue while WMMAs run.
// ---------------------------------------------------------------------------
template <int BT, int GELU, int FULL>
__global__ __launch_bounds__(256) void k_gemm(const float* __restrict__ A,
                                              const float* __restrict__ W,
                                              const float* __restrict__ Bias,
                                              float* __restrict__ C,
                                              int M, int N, int K) {
    __shared__ __bf16 sA[128 * 40];   // row-major, stride 40 halves
    __shared__ __bf16 sB[128 * 40];   // column-major: sB[n*40 + k]

    const int tid  = threadIdx.x;
    const int wave = tid >> 5;
    const int lane = tid & 31;
    const int wm   = wave >> 2;       // 0..1
    const int wn   = wave & 3;        // 0..3
    const int bm0  = blockIdx.y * 128;
    const int bn0  = blockIdx.x * 128;

    v8f acc[4][2];
    const v8f zero = {0.f, 0.f, 0.f, 0.f, 0.f, 0.f, 0.f, 0.f};
#pragma unroll
    for (int mt = 0; mt < 4; ++mt)
#pragma unroll
        for (int nt = 0; nt < 2; ++nt) acc[mt][nt] = zero;

    // A staging coords: thread -> (row, 16-col segment)
    const int ar = tid >> 1, ac = (tid & 1) * 16;
    const bool aok = FULL || ((bm0 + ar) < M);
    const float* aptr = A + (long)(bm0 + ar) * K + ac;

    // B staging coords
    const int bn = tid >> 1, kc = (tid & 1) * 16;          // BT=1 (row-major in K)
    const int kp = (tid >> 4) * 2, nb = (tid & 15) * 8;    // BT=0 (pair of K rows x 8 cols)
    const bool wok = FULL || (BT ? ((bn0 + bn) < N) : true);
    const float* wptr = BT ? (W + (long)(bn0 + bn) * K + kc)
                           : (W + (long)kp * N + bn0 + nb);

    float ra[16], rw[16];
    auto loadA = [&](int k0) {
#pragma unroll
        for (int j = 0; j < 16; ++j) ra[j] = aok ? aptr[k0 + j] : 0.f;
    };
    auto loadB = [&](int k0) {
        if (BT) {
#pragma unroll
            for (int j = 0; j < 16; ++j) rw[j] = wok ? wptr[k0 + j] : 0.f;
        } else {
#pragma unroll
            for (int j = 0; j < 8; ++j)
                rw[j] = (FULL || (bn0 + nb + j) < N) ? wptr[(long)k0 * N + j] : 0.f;
#pragma unroll
            for (int j = 0; j < 8; ++j)
                rw[8 + j] = (FULL || (bn0 + nb + j) < N) ? wptr[(long)k0 * N + N + j] : 0.f;
        }
    };
    auto storeTiles = [&]() {
#pragma unroll
        for (int j = 0; j < 16; j += 2)
            st_bf2(&sA[ar * 40 + ac + j], ra[j], ra[j + 1]);
        if (BT) {
#pragma unroll
            for (int j = 0; j < 16; j += 2)
                st_bf2(&sB[bn * 40 + kc + j], rw[j], rw[j + 1]);
        } else {
#pragma unroll
            for (int j = 0; j < 8; ++j)   // (k=kp, k=kp+1) for column nb+j
                st_bf2(&sB[(nb + j) * 40 + kp], rw[j], rw[8 + j]);
        }
    };

    const int row = lane & 15;
    const int kg  = (lane >> 4) * 8;     // A-frag K sub-offset (0 or 8)
    const int kb0 = (lane >> 4) * 16;    // 0 or 16; unused, B-frag also reads at kg

    loadA(0);
    loadB(0);
    for (int k0 = 0; k0 < K; k0 += 32) {
        storeTiles();
        __syncthreads();
        if (k0 + 32 < K) {               // prefetch next tile during WMMA
            loadA(k0 + 32);
            loadB(k0 + 32);
        }
        v16bf afr[4], bfr[2];
#pragma unroll
        for (int mt = 0; mt < 4; ++mt) {
            const int r = wm * 64 + mt * 16 + row;
            afr[mt] = frag16(&sA[r * 40 + kg], &sA[r * 40 + kg + 16]);
        }
#pragma unroll
        for (int nt = 0; nt < 2; ++nt) {
            const int cidx = wn * 32 + nt * 16 + row;
            bfr[nt] = frag16(&sB[cidx * 40 + kg], &sB[cidx * 40 + kg + 16]);
        }
#pragma unroll
        for (int mt = 0; mt < 4; ++mt)
#pragma unroll
            for (int nt = 0; nt < 2; ++nt)
                acc[mt][nt] = __builtin_amdgcn_wmma_f32_16x16x32_bf16(
                    false, afr[mt], false, bfr[nt], (short)0, acc[mt][nt], false, false);
        __syncthreads();
    }

    // ---- epilogue
    const int cc = lane & 15;
    const int rb = (lane >> 4) * 8;
#pragma unroll
    for (int mt = 0; mt < 4; ++mt)
#pragma unroll
        for (int nt = 0; nt < 2; ++nt)
#pragma unroll
            for (int i = 0; i < 8; ++i) {
                const int gm = bm0 + wm * 64 + mt * 16 + rb + i;
                const int gn = bn0 + wn * 32 + nt * 16 + cc;
                if (FULL || (gm < M && gn < N)) {
                    float v = acc[mt][nt][i] + Bias[gn];
                    if (GELU) v = 0.5f * v * (1.0f + erff(v * 0.70710678118f));
                    C[(long)gm * N + gn] = v;
                }
            }
}

// ---------------------------------------------------------------------------
// Embedding: out[row] = tok_emb[ids[row]] + pos[row % poslen]
// ---------------------------------------------------------------------------
__global__ __launch_bounds__(256) void k_embed(const int* __restrict__ ids,
                                               const float* __restrict__ emb,
                                               const float* __restrict__ pos,
                                               float* __restrict__ out, int poslen) {
    const int row = blockIdx.x, tid = threadIdx.x;
    const int id = ids[row];
    const int pr = row % poslen;
#pragma unroll
    for (int j = 0; j < 3; ++j) {
        const int idx = tid + j * 256;
        out[(long)row * Dm + idx] = emb[(long)id * Dm + idx] + pos[(long)pr * Dm + idx];
    }
}

// ---------------------------------------------------------------------------
// LayerNorm over D=768 (optionally with residual add): out = LN(in + res)
// ---------------------------------------------------------------------------
__global__ __launch_bounds__(256) void k_ln(const float* __restrict__ in,
                                            const float* __restrict__ res,
                                            const float* __restrict__ g,
                                            const float* __restrict__ bta,
                                            float* __restrict__ out) {
    const int row = blockIdx.x, tid = threadIdx.x;
    __shared__ float red[256];
    float vals[3];
    float s = 0.f;
#pragma unroll
    for (int j = 0; j < 3; ++j) {
        const int idx = tid + j * 256;
        float x = in[(long)row * Dm + idx];
        if (res) x += res[(long)row * Dm + idx];
        vals[j] = x;
        s += x;
    }
    red[tid] = s;
    __syncthreads();
    for (int st = 128; st > 0; st >>= 1) { if (tid < st) red[tid] += red[tid + st]; __syncthreads(); }
    const float mean = red[0] * (1.f / 768.f);
    __syncthreads();
    float s2 = 0.f;
#pragma unroll
    for (int j = 0; j < 3; ++j) { const float d = vals[j] - mean; s2 += d * d; }
    red[tid] = s2;
    __syncthreads();
    for (int st = 128; st > 0; st >>= 1) { if (tid < st) red[tid] += red[tid + st]; __syncthreads(); }
    const float var = red[0] * (1.f / 768.f);
    const float rstd = rsqrtf(var + 1e-5f);
#pragma unroll
    for (int j = 0; j < 3; ++j) {
        const int idx = tid + j * 256;
        out[(long)row * Dm + idx] = (vals[j] - mean) * rstd * g[idx] + bta[idx];
    }
}

// ---------------------------------------------------------------------------
// Encoder banded local+global attention (Longformer).
// Block = (b, h, chunk n). Thread = one query (256). Window = chunks
// n-1..n+1 staged via async global->LDS in 64-key fp32 sub-blocks.
// Online softmax seeded with the (unmasked) global-token-0 score.
// q/k/v layout: [B, S, H*DH] row-major.
// ---------------------------------------------------------------------------
__global__ __launch_bounds__(256) void k_enc_attn(const float* __restrict__ q,
                                                  const float* __restrict__ k,
                                                  const float* __restrict__ v,
                                                  const int* __restrict__ mask,
                                                  float* __restrict__ out) {
    __shared__ float sK[64][68];
    __shared__ float sV[64][68];
    const int bid = blockIdx.x;
    const int n = bid & 15;
    const int h = (bid >> 4) % Hh;
    const int b = bid / (16 * Hh);
    const int tid = threadIdx.x;
    const int qidx = n * Ww + tid;
    const long base = ((long)b * Ss) * Dm + h * DH;

    float qv[DH];
    {
        const float* qp = q + base + (long)qidx * Dm;
#pragma unroll
        for (int d = 0; d < DH; ++d) qv[d] = qp[d] * 0.125f;
    }
    // global key 0 (unmasked per reference)
    float m, l, acc[DH];
    {
        const float* k0 = k + base;
        const float* v0 = v + base;
        float s = 0.f;
#pragma unroll
        for (int d = 0; d < DH; ++d) s += qv[d] * k0[d];
        m = s; l = 1.f;
#pragma unroll
        for (int d = 0; d < DH; ++d) acc[d] = v0[d];
    }

    for (int c = n - 1; c <= n + 1; ++c) {
        if (c < 0 || c >= NC) continue;
        for (int sb = 0; sb < 4; ++sb) {
            const int kbase = c * Ww + sb * 64;
            __syncthreads();
            {   // async-stage 64 keys/values: thread -> row tid>>2, segment tid&3
                const int r = tid >> 2, sg = (tid & 3) * 16;
                const float* kp = k + base + (long)(kbase + r) * Dm + sg;
                const float* vp = v + base + (long)(kbase + r) * Dm + sg;
#pragma unroll
                for (int j = 0; j < 4; ++j) {
                    async_b128(&sK[r][sg + j * 4], kp + j * 4);
                    async_b128(&sV[r][sg + j * 4], vp + j * 4);
                }
            }
            async_wait0();
            __syncthreads();
            for (int j = 0; j < 64; ++j) {
                const int kpos = kbase + j;
                const int rel = kpos - qidx;
                if (kpos == 0 || rel > Ww || rel < -Ww) continue;
                if (mask[b * Ss + kpos] == 0) continue;
                float s = 0.f;
#pragma unroll
                for (int d = 0; d < DH; ++d) s += qv[d] * sK[j][d];
                if (s > m) {
                    const float cf = __expf(m - s);
                    l *= cf;
#pragma unroll
                    for (int d = 0; d < DH; ++d) acc[d] *= cf;
                    m = s;
                }
                const float p = __expf(s - m);
                l += p;
#pragma unroll
                for (int d = 0; d < DH; ++d) acc[d] += p * sV[j][d];
            }
        }
    }
    const float inv = 1.f / l;
    float* op = out + base + (long)qidx * Dm;
#pragma unroll
    for (int d = 0; d < DH; ++d) op[d] = acc[d] * inv;
}

// Dense attention for the single global query (position 0) per (b,h).
__global__ __launch_bounds__(256) void k_enc_attn_g(const float* __restrict__ q,
                                                    const float* __restrict__ k,
                                                    const float* __restrict__ v,
                                                    const int* __restrict__ mask,
                                                    float* __restrict__ out) {
    const int h = blockIdx.x % Hh, b = blockIdx.x / Hh;
    const long base = ((long)b * Ss) * Dm + h * DH;
    __shared__ float q0[DH];
    __shared__ float red[256];
    __shared__ float accS[DH];
    __shared__ float lS;
    const int tid = threadIdx.x;
    if (tid < DH) { q0[tid] = q[base + tid] * 0.125f; accS[tid] = 0.f; }
    if (tid == 0) lS = 0.f;
    __syncthreads();

    float m = -1e30f;
    for (int i = tid; i < Ss; i += 256) {
        if (mask[b * Ss + i] == 0) continue;
        const float* kp = k + base + (long)i * Dm;
        float s = 0.f;
#pragma unroll
        for (int d = 0; d < DH; ++d) s += q0[d] * kp[d];
        m = fmaxf(m, s);
    }
    red[tid] = m;
    __syncthreads();
    for (int st = 128; st > 0; st >>= 1) { if (tid < st) red[tid] = fmaxf(red[tid], red[tid + st]); __syncthreads(); }
    m = red[0];
    __syncthreads();

    float lsum = 0.f, lacc[DH];
#pragma unroll
    for (int d = 0; d < DH; ++d) lacc[d] = 0.f;
    for (int i = tid; i < Ss; i += 256) {
        if (mask[b * Ss + i] == 0) continue;
        const float* kp = k + base + (long)i * Dm;
        const float* vp = v + base + (long)i * Dm;
        float s = 0.f;
#pragma unroll
        for (int d = 0; d < DH; ++d) s += q0[d] * kp[d];
        const float p = __expf(s - m);
        lsum += p;
#pragma unroll
        for (int d = 0; d < DH; ++d) lacc[d] += p * vp[d];
    }
#pragma unroll
    for (int d = 0; d < DH; ++d) atomicAdd(&accS[d], lacc[d]);
    atomicAdd(&lS, lsum);
    __syncthreads();
    if (tid < DH) out[base + tid] = accS[tid] / lS;
}

// ---------------------------------------------------------------------------
// Decoder causal self-attention. Block = (b,h), thread = query t (128).
// ---------------------------------------------------------------------------
__global__ __launch_bounds__(128) void k_dec_self(const float* __restrict__ q,
                                                  const float* __restrict__ k,
                                                  const float* __restrict__ v,
                                                  float* __restrict__ out) {
    __shared__ float sK[64][68];
    __shared__ float sV[64][68];
    const int h = blockIdx.x % Hh, b = blockIdx.x / Hh;
    const long base = ((long)b * Tt) * Dm + h * DH;
    const int t = threadIdx.x;

    float qv[DH];
    {
        const float* qp = q + base + (long)t * Dm;
#pragma unroll
        for (int d = 0; d < DH; ++d) qv[d] = qp[d] * 0.125f;
    }
    float m = -1e30f, l = 0.f, acc[DH];
#pragma unroll
    for (int d = 0; d < DH; ++d) acc[d] = 0.f;

    for (int sb = 0; sb < 2; ++sb) {
        __syncthreads();
        {
            const int r = t >> 1, sg = (t & 1) * 32;
            const float* kp = k + base + (long)(sb * 64 + r) * Dm + sg;
            const float* vp = v + base + (long)(sb * 64 + r) * Dm + sg;
#pragma unroll
            for (int j = 0; j < 8; ++j) {
                async_b128(&sK[r][sg + j * 4], kp + j * 4);
                async_b128(&sV[r][sg + j * 4], vp + j * 4);
            }
        }
        async_wait0();
        __syncthreads();
        for (int j = 0; j < 64; ++j) {
            const int kpos = sb * 64 + j;
            if (kpos > t) continue;   // causal
            float s = 0.f;
#pragma unroll
            for (int d = 0; d < DH; ++d) s += qv[d] * sK[j][d];
            if (s > m) {
                const float cf = __expf(m - s);
                l *= cf;
#pragma unroll
                for (int d = 0; d < DH; ++d) acc[d] *= cf;
                m = s;
            }
            const float p = __expf(s - m);
            l += p;
#pragma unroll
            for (int d = 0; d < DH; ++d) acc[d] += p * sV[j][d];
        }
    }
    const float inv = 1.f / l;
    float* op = out + base + (long)t * Dm;
#pragma unroll
    for (int d = 0; d < DH; ++d) op[d] = acc[d] * inv;
}

// ---------------------------------------------------------------------------
// Cross attention: 128 decoder queries over S=4096 encoder keys (masked).
// ---------------------------------------------------------------------------
__global__ __launch_bounds__(128) void k_cross(const float* __restrict__ q,
                                               const float* __restrict__ k,
                                               const float* __restrict__ v,
                                               const int* __restrict__ mask,
                                               float* __restrict__ out) {
    __shared__ float sK[64][68];
    __shared__ float sV[64][68];
    const int h = blockIdx.x % Hh, b = blockIdx.x / Hh;
    const long baseQ = ((long)b * Tt) * Dm + h * DH;
    const long baseK = ((long)b * Ss) * Dm + h * DH;
    const int t = threadIdx.x;

    float qv[DH];
    {
        const float* qp = q + baseQ + (long)t * Dm;
#pragma unroll
        for (int d = 0; d < DH; ++d) qv[d] = qp[d] * 0.125f;
    }
    float m = -1e30f, l = 0.f, acc[DH];
#pragma unroll
    for (int d = 0; d < DH; ++d) acc[d] = 0.f;

    for (int blk = 0; blk < Ss / 64; ++blk) {
        __syncthreads();
        {
            const int r = t >> 1, sg = (t & 1) * 32;
            const float* kp = k + baseK + (long)(blk * 64 + r) * Dm + sg;
            const float* vp = v + baseK + (long)(blk * 64 + r) * Dm + sg;
#pragma unroll
            for (int j = 0; j < 8; ++j) {
                async_b128(&sK[r][sg + j * 4], kp + j * 4);
                async_b128(&sV[r][sg + j * 4], vp + j * 4);
            }
        }
        async_wait0();
        __syncthreads();
        for (int j = 0; j < 64; ++j) {
            const int kpos = blk * 64 + j;
            if (mask[b * Ss + kpos] == 0) continue;
            float s = 0.f;
#pragma unroll
            for (int d = 0; d < DH; ++d) s += qv[d] * sK[j][d];
            if (s > m) {
                const float cf = __expf(m - s);
                l *= cf;
#pragma unroll
                for (int d = 0; d < DH; ++d) acc[d] *= cf;
                m = s;
            }
            const float p = __expf(s - m);
            l += p;
#pragma unroll
            for (int d = 0; d < DH; ++d) acc[d] += p * sV[j][d];
        }
    }
    const float inv = 1.f / fmaxf(l, 1e-30f);
    float* op = out + baseQ + (long)t * Dm;
#pragma unroll
    for (int d = 0; d < DH; ++d) op[d] = acc[d] * inv;
}

// ---------------------------------------------------------------------------
// dec_ids = concat(trg[ne], trg[:-1]) where ne = count(trg != PAD) - 1
// ---------------------------------------------------------------------------
__global__ __launch_bounds__(128) void k_dec_ids(const int* __restrict__ trg,
                                                 int* __restrict__ dec) {
    const int b = blockIdx.x, tid = threadIdx.x;
    __shared__ int red[128];
    red[tid] = (trg[b * Tt + tid] != PADID) ? 1 : 0;
    __syncthreads();
    for (int st = 64; st > 0; st >>= 1) { if (tid < st) red[tid] += red[tid + st]; __syncthreads(); }
    const int ne = max(red[0] - 1, 0);
    if (tid == 0) dec[b * Tt] = trg[b * Tt + ne];
    else dec[b * Tt + tid] = trg[b * Tt + tid - 1];
}

// ---------------------------------------------------------------------------
// Per-row log-softmax + NLL over V logits. rowout[row] = nll*valid,
// rowout[256+row] = valid.
// ---------------------------------------------------------------------------
__global__ __launch_bounds__(256) void k_nll(const float* __restrict__ logits,
                                             const int* __restrict__ trg,
                                             float* __restrict__ rowout) {
    const int row = blockIdx.x, tid = threadIdx.x;
    const float* lr = logits + (long)row * Vv;
    __shared__ float red[256];
    float m = -1e30f;
    for (int i = tid; i < Vv; i += 256) m = fmaxf(m, lr[i]);
    red[tid] = m;
    __syncthreads();
    for (int st = 128; st > 0; st >>= 1) { if (tid < st) red[tid] = fmaxf(red[tid], red[tid + st]); __syncthreads(); }
    m = red[0];
    __syncthreads();
    float s = 0.f;
    for (int i = tid; i < Vv; i += 256) s += __expf(lr[i] - m);
    red[tid] = s;
    __syncthreads();
    for (int st = 128; st > 0; st >>= 1) { if (tid < st) red[tid] += red[tid + st]; __syncthreads(); }
    if (tid == 0) {
        const float l = red[0];
        const int tgt = trg[row];
        const float lp = lr[tgt] - m - logf(l);
        const float valid = (tgt != PADID) ? 1.f : 0.f;
        rowout[row] = -lp * valid;
        rowout[256 + row] = valid;
    }
}

__global__ __launch_bounds__(256) void k_loss(const float* __restrict__ rowout,
                                              float* __restrict__ out) {
    const int tid = threadIdx.x;
    __shared__ float red[256];
    red[tid] = rowout[tid];
    __syncthreads();
    for (int st = 128; st > 0; st >>= 1) { if (tid < st) red[tid] += red[tid + st]; __syncthreads(); }
    const float sn = red[0];
    __syncthreads();
    red[tid] = rowout[256 + tid];
    __syncthreads();
    for (int st = 128; st > 0; st >>= 1) { if (tid < st) red[tid] += red[tid + st]; __syncthreads(); }
    if (tid == 0) out[42] = sn / fmaxf(red[0], 1.f);
}

// ---------------------------------------------------------------------------
// Classification heads from enc_h[:,0]: score(2), conf(20), rate(20).
// ---------------------------------------------------------------------------
__global__ __launch_bounds__(64) void k_heads(const float* __restrict__ ench,
                                              const float* __restrict__ sw, const float* __restrict__ sb,
                                              const float* __restrict__ cw, const float* __restrict__ cb,
                                              const float* __restrict__ rw, const float* __restrict__ rb,
                                              float* __restrict__ out) {
    const int tid = threadIdx.x;
    if (tid >= 42) return;
    const int b = tid / 21, r = tid % 21;
    const float* cls = ench + (long)b * Ss * Dm;
    if (r == 0) {
        float a = sb[0];
        for (int i = 0; i < Dm; ++i) a += cls[i] * sw[i];
        out[b] = 1.f / (1.f + __expf(-a));
    } else if (r <= 10) {
        const int j = r - 1;
        float a = cb[j];
        for (int i = 0; i < Dm; ++i) a += cls[i] * cw[i * 10 + j];
        out[2 + b * 10 + j] = a;
    } else {
        const int j = r - 11;
        float a = rb[j];
        for (int i = 0; i < Dm; ++i) a += cls[i] * rw[i * 10 + j];
        out[22 + b * 10 + j] = 1.f / (1.f + __expf(-a));
    }
}

// ---------------------------------------------------------------------------
// Host driver
// ---------------------------------------------------------------------------
extern "C" void kernel_launch(void* const* d_in, const int* in_sizes, int n_in,
                              void* d_out, int out_size, void* d_ws, size_t ws_size,
                              hipStream_t stream) {
    (void)in_sizes; (void)n_in; (void)out_size; (void)ws_size;
    const int* input_ids = (const int*)d_in[0];
    const int* amask     = (const int*)d_in[1];
    const int* trg       = (const int*)d_in[2];
    auto PF = [&](int i) -> const float* { return (const float*)d_in[i]; };
    // param index map (depth-first flatten of the params dict)
    // 3 tok_emb, 4 enc_pos, 5 dec_pos, 6/7 enc_emb_ln, 8/9 dec_emb_ln,
    // 10+16*L enc layers, 42+26*L dec layers, 94 final_bias, 95..100 heads.

    // ---- workspace carve
    char* p = (char*)d_ws;
    auto alloc = [&](size_t bytes) -> void* {
        void* r = (void*)p;
        p += (bytes + 255) & ~(size_t)255;
        return r;
    };
    const size_t actB = (size_t)2 * Ss * Dm * sizeof(float);       // 25.2 MB
    float* X   = (float*)alloc(actB);                              // enc act / enc_h
    float* Qb  = (float*)alloc(actB);
    float* Kb  = (float*)alloc(actB);
    float* Vb  = (float*)alloc(actB);
    float* T0  = (float*)alloc(actB);
    float* HID = (float*)alloc((size_t)2 * Ss * FFm * sizeof(float)); // 100.7 MB
    float* LOGITS = HID;                                           // reuse after encoder
    const size_t dactB = (size_t)2 * Tt * Dm * sizeof(float);
    float* Y   = (float*)alloc(dactB);
    float* YQ  = (float*)alloc(dactB);
    float* YK  = (float*)alloc(dactB);
    float* YV  = (float*)alloc(dactB);
    float* YT  = (float*)alloc(dactB);
    float* DHID = (float*)alloc((size_t)2 * Tt * FFm * sizeof(float));
    int*   DEC_IDS = (int*)alloc(2 * Tt * sizeof(int));
    float* ROW = (float*)alloc(512 * sizeof(float));
    float* OUT = (float*)d_out;

    auto gemm = [&](const float* A, const float* W, const float* B, float* C,
                    int M, int N, int K, int bt, int gelu) {
        dim3 g((N + 127) / 128, (M + 127) / 128);
        if (bt == 1)        k_gemm<1, 0, 0><<<g, 256, 0, stream>>>(A, W, B, C, M, N, K);
        else if (gelu == 1) k_gemm<0, 1, 1><<<g, 256, 0, stream>>>(A, W, B, C, M, N, K);
        else                k_gemm<0, 0, 1><<<g, 256, 0, stream>>>(A, W, B, C, M, N, K);
    };

    // ================= Encoder =================
    k_embed<<<2 * Ss, 256, 0, stream>>>(input_ids, PF(3), PF(4), X, Ss);
    k_ln<<<2 * Ss, 256, 0, stream>>>(X, nullptr, PF(6), PF(7), X);
    for (int L = 0; L < 2; ++L) {
        const int pb = 10 + L * 16;
        gemm(X, PF(pb + 0), PF(pb + 1), Qb, 2 * Ss, Dm, Dm, 0, 0);
        gemm(X, PF(pb + 2), PF(pb + 3), Kb, 2 * Ss, Dm, Dm, 0, 0);
        gemm(X, PF(pb + 4), PF(pb + 5), Vb, 2 * Ss, Dm, Dm, 0, 0);
        k_enc_attn<<<2 * Hh * NC, 256, 0, stream>>>(Qb, Kb, Vb, amask, T0);
        k_enc_attn_g<<<2 * Hh, 256, 0, stream>>>(Qb, Kb, Vb, amask, T0);
        gemm(T0, PF(pb + 6), PF(pb + 7), Qb, 2 * Ss, Dm, Dm, 0, 0);
        k_ln<<<2 * Ss, 256, 0, stream>>>(Qb, X, PF(pb + 8), PF(pb + 9), X);
        gemm(X, PF(pb + 10), PF(pb + 11), HID, 2 * Ss, FFm, Dm, 0, 1);   // fused GELU
        gemm(HID, PF(pb + 12), PF(pb + 13), T0, 2 * Ss, Dm, FFm, 0, 0);
        k_ln<<<2 * Ss, 256, 0, stream>>>(T0, X, PF(pb + 14), PF(pb + 15), X);
    }
    // X now holds enc_h

    // ================= Decoder =================
    k_dec_ids<<<2, 128, 0, stream>>>(trg, DEC_IDS);
    k_embed<<<2 * Tt, 256, 0, stream>>>(DEC_IDS, PF(3), PF(5), Y, Tt);
    k_ln<<<2 * Tt, 256, 0, stream>>>(Y, nullptr, PF(8), PF(9), Y);
    for (int L = 0; L < 2; ++L) {
        const int pb = 42 + L * 26;
        // self-attention (causal)
        gemm(Y, PF(pb + 0), PF(pb + 1), YQ, 2 * Tt, Dm, Dm, 0, 0);
        gemm(Y, PF(pb + 2), PF(pb + 3), YK, 2 * Tt, Dm, Dm, 0, 0);
        gemm(Y, PF(pb + 4), PF(pb + 5), YV, 2 * Tt, Dm, Dm, 0, 0);
        k_dec_self<<<2 * Hh, 128, 0, stream>>>(YQ, YK, YV, YT);
        gemm(YT, PF(pb + 6), PF(pb + 7), YQ, 2 * Tt, Dm, Dm, 0, 0);
        k_ln<<<2 * Tt, 256, 0, stream>>>(YQ, Y, PF(pb + 8), PF(pb + 9), Y);
        // cross-attention
        gemm(Y, PF(pb + 16), PF(pb + 17), YQ, 2 * Tt, Dm, Dm, 0, 0);
        gemm(X, PF(pb + 18), PF(pb + 19), Kb, 2 * Ss, Dm, Dm, 0, 0);
        gemm(X, PF(pb + 20), PF(pb + 21), Vb, 2 * Ss, Dm, Dm, 0, 0);
        k_cross<<<2 * Hh, 128, 0, stream>>>(YQ, Kb, Vb, amask, YT);
        gemm(YT, PF(pb + 22), PF(pb + 23), YQ, 2 * Tt, Dm, Dm, 0, 0);
        k_ln<<<2 * Tt, 256, 0, stream>>>(YQ, Y, PF(pb + 24), PF(pb + 25), Y);
        // FFN
        gemm(Y, PF(pb + 10), PF(pb + 11), DHID, 2 * Tt, FFm, Dm, 0, 1);
        gemm(DHID, PF(pb + 12), PF(pb + 13), YT, 2 * Tt, Dm, FFm, 0, 0);
        k_ln<<<2 * Tt, 256, 0, stream>>>(YT, Y, PF(pb + 14), PF(pb + 15), Y);
    }

    // ================= LM head + loss =================
    gemm(Y, PF(3), PF(94), LOGITS, 2 * Tt, Vv, Dm, 1, 0);   // tok_emb as [N,K]
    k_nll<<<2 * Tt, 256, 0, stream>>>(LOGITS, trg, ROW);
    k_loss<<<1, 256, 0, stream>>>(ROW, OUT);

    // ================= heads =================
    k_heads<<<1, 64, 0, stream>>>(X, PF(95), PF(96), PF(97), PF(98), PF(99), PF(100), OUT);
}
